// MultiInputLSTMCell_45689862095431
// MI455X (gfx1250) — matrix operand order repaired and checked by reading, not compiled, not run on hardware
//
#include <hip/hip_runtime.h>

// ---------------------------------------------------------------------------
// MultiInputLSTMCell for MI455X (gfx1250, wave32, WMMA)
//
// Roofline: dominant GEMM  c_input[4096x2048] @ aW_hh[2048x2048]^T = 34.4 GFLOP.
// fp32 matrix path (16x16x4) would be compute-bound; bf16 WMMA (16x16x32)
// pushes it under the HBM floor (~285 MB total traffic / 23.3 TB/s ~ 12 us).
// GEMVs (W_ih/W_hh/aW_ih) are pure memory streams -> plain float4 VALU dot.
// Softmax-over-(C+1) is fused into the GEMM epilogue with deterministic
// per-tile partial sums (no float atomics).
// ---------------------------------------------------------------------------

#define IN_DIM 4096
#define H_DIM  2048
#define C_DIM  4096
#define KDIM   H_DIM          // GEMM reduction dim
#define MT_PER_WAVE 4         // 4 x 16 = 64 rows per wave
#define NT_PER_WAVE 4         // 4 x 16 = 64 cols per wave
#define M_TILES (C_DIM / 64)  // 64 row-groups for partial reduction

typedef __bf16 v16bf __attribute__((ext_vector_type(16)));
typedef __bf16 v8bf  __attribute__((ext_vector_type(8)));
typedef float  v8f   __attribute__((ext_vector_type(8)));

__device__ __forceinline__ unsigned int f2bf_rne(float f) {
  unsigned int u = __float_as_uint(f);
  return (u + 0x7FFFu + ((u >> 16) & 1u)) >> 16;   // round-to-nearest-even
}
__device__ __forceinline__ float sigmoid_fast(float x) {
  return 1.0f / (1.0f + __expf(-x));
}

// --- fp32 -> bf16 conversion of c_input (C x K) and aW_hh (H x K), packed
//     back-to-back in one bf16 array ------------------------------------------
__global__ void __launch_bounds__(256) k_convert_bf16(
    const float* __restrict__ c_input, const float* __restrict__ aW_hh,
    unsigned int* __restrict__ dst /* packed 2 x bf16 per uint */) {
  const int CH4 = (C_DIM * KDIM) / 4;   // float4 count in c_input
  const int WH4 = (H_DIM * KDIM) / 4;   // float4 count in aW_hh
  int idx = blockIdx.x * 256 + threadIdx.x;
  if (idx >= CH4 + WH4) return;
  float4 v = (idx < CH4) ? ((const float4*)c_input)[idx]
                         : ((const float4*)aW_hh)[idx - CH4];
  uint2 p;
  p.x = f2bf_rne(v.x) | (f2bf_rne(v.y) << 16);
  p.y = f2bf_rne(v.z) | (f2bf_rne(v.w) << 16);
  ((uint2*)dst)[idx] = p;
}

// --- LSTM gate GEMV: rows for i (0..H), o (2H..3H), g (3H..4H); f is dead ----
__global__ void __launch_bounds__(256) k_gates_gemv(
    const float* __restrict__ W_ih, const float* __restrict__ b_ih,
    const float* __restrict__ W_hh, const float* __restrict__ b_hh,
    const float* __restrict__ x, const float* __restrict__ h0,
    float* __restrict__ gates_pre /* [3*H]: i, o, g pre-activations */) {
  int wid  = (blockIdx.x * blockDim.x + threadIdx.x) >> 5;  // wave per row
  int lane = threadIdx.x & 31;
  int blk  = wid >> 11;                 // 0=i, 1=o, 2=g
  int hrow = wid & (H_DIM - 1);
  int row  = hrow + (blk == 0 ? 0 : (blk == 1 ? 2 * H_DIM : 3 * H_DIM));

  const float4* wa = (const float4*)(W_ih + (size_t)row * IN_DIM);
  const float4* xv = (const float4*)x;
  float s = 0.0f;
#pragma unroll 4
  for (int i = lane; i < IN_DIM / 4; i += 32) {
    float4 w = wa[i], v = xv[i];
    s = fmaf(w.x, v.x, s); s = fmaf(w.y, v.y, s);
    s = fmaf(w.z, v.z, s); s = fmaf(w.w, v.w, s);
  }
  const float4* wh = (const float4*)(W_hh + (size_t)row * H_DIM);
  const float4* hv = (const float4*)h0;
#pragma unroll 4
  for (int i = lane; i < H_DIM / 4; i += 32) {
    float4 w = wh[i], v = hv[i];
    s = fmaf(w.x, v.x, s); s = fmaf(w.y, v.y, s);
    s = fmaf(w.z, v.z, s); s = fmaf(w.w, v.w, s);
  }
  for (int off = 16; off > 0; off >>= 1) s += __shfl_down(s, off, 32);
  if (lane == 0) gates_pre[wid] = s + b_ih[row] + b_hh[row];
}

// --- ax[h] = x @ aW_ih.T + ab_ih + ab_hh  (biases folded in) -----------------
__global__ void __launch_bounds__(256) k_ax_gemv(
    const float* __restrict__ aW_ih, const float* __restrict__ ab_ih,
    const float* __restrict__ ab_hh, const float* __restrict__ x,
    float* __restrict__ ax) {
  int wid  = (blockIdx.x * blockDim.x + threadIdx.x) >> 5;
  int lane = threadIdx.x & 31;
  const float4* wa = (const float4*)(aW_ih + (size_t)wid * IN_DIM);
  const float4* xv = (const float4*)x;
  float s = 0.0f;
#pragma unroll 4
  for (int i = lane; i < IN_DIM / 4; i += 32) {
    float4 w = wa[i], v = xv[i];
    s = fmaf(w.x, v.x, s); s = fmaf(w.y, v.y, s);
    s = fmaf(w.z, v.z, s); s = fmaf(w.w, v.w, s);
  }
  for (int off = 16; off > 0; off >>= 1) s += __shfl_down(s, off, 32);
  if (lane == 0) ax[wid] = s + ab_ih[wid] + ab_hh[wid];
}

// --- bf16 WMMA GEMM (alpha = c_input @ aW_hh^T) fused with sigmoid/exp and
//     the column-wise softmax partial reduction --------------------------------
// Each wave: 64x64 output tile = 4x4 V_WMMA_F32_16X16X32_BF16 accumulators.
// A layout (B alike, contiguous K): lane<16 -> M=lane, K(+0..15); lane>=16 -> K(+16..31).
// C/D layout: VGPR j -> M = j (+8 for lanes 16..31), lane%16 -> N.
__global__ void __launch_bounds__(256) k_alpha_wmma(
    const __bf16* __restrict__ Abf,      // c_input bf16 [C, K]
    const __bf16* __restrict__ Bbf,      // aW_hh  bf16 [H, K]
    const float*  __restrict__ c_input,  // fp32  [C, H] for the weighted sum
    const float*  __restrict__ ax,       // [H]
    float* __restrict__ denP,            // [M_TILES, H] partial sum(exp(alpha))
    float* __restrict__ numP) {          // [M_TILES, H] partial sum(exp*c_in)
  int wid  = (blockIdx.x * blockDim.x + threadIdx.x) >> 5;
  int lane = threadIdx.x & 31;
  int tm   = wid & (M_TILES - 1);        // 0..63 row-tile (64 rows)
  int tn   = wid >> 6;                   // 0..31 col-tile (64 cols)
  int row0 = tm * 64;
  int col0 = tn * 64;
  int sel  = lane >> 4;                  // K-half select
  int lr   = lane & 15;

  v8f acc[MT_PER_WAVE][NT_PER_WAVE] = {};

  for (int k0 = 0; k0 < KDIM; k0 += 32) {
    v16bf bfrag[NT_PER_WAVE];
#pragma unroll
    for (int nt = 0; nt < NT_PER_WAVE; ++nt) {
      int n = col0 + nt * 16 + lr;       // B row n == output column n
      bfrag[nt] = *(const v16bf*)(Bbf + (size_t)n * KDIM + k0 + sel * 16);
    }
#pragma unroll
    for (int mt = 0; mt < MT_PER_WAVE; ++mt) {
      int m = row0 + mt * 16 + lr;
      const __bf16* ap = Abf + (size_t)m * KDIM + k0 + sel * 16;
      v8bf alo = *(const v8bf*)ap;            // K +0..7   / +16..23
      v8bf ahi = *(const v8bf*)(ap + 8);      // K +8..15  / +24..31
      v16bf afrag = __builtin_shufflevector(
          alo, ahi, 0, 1, 2, 3, 4, 5, 6, 7, 8, 9, 10, 11, 12, 13, 14, 15);
#pragma unroll
      for (int nt = 0; nt < NT_PER_WAVE; ++nt) {
        acc[mt][nt] = __builtin_amdgcn_wmma_f32_16x16x32_bf16(
            false, afrag, false, bfrag[nt], (short)0, acc[mt][nt],
            false, false);
      }
    }
  }

  // Epilogue: alpha = sigmoid(acc + ax); accumulate exp(alpha) and
  // exp(alpha)*c_input per column h; fold lane pairs (L, L+16) then write
  // this wave's unique (tm, h) partial slots (deterministic, no atomics).
#pragma unroll
  for (int nt = 0; nt < NT_PER_WAVE; ++nt) {
    int h = col0 + nt * 16 + lr;
    float axh = ax[h];
    float dsum = 0.0f, nsum = 0.0f;
#pragma unroll
    for (int mt = 0; mt < MT_PER_WAVE; ++mt) {
#pragma unroll
      for (int j = 0; j < 8; ++j) {
        int m = row0 + mt * 16 + j + sel * 8;
        float a = sigmoid_fast(acc[mt][nt][j] + axh);
        float e = __expf(a);
        dsum += e;
        nsum += e * c_input[(size_t)m * H_DIM + h];
      }
    }
    dsum += __shfl_xor(dsum, 16, 32);
    nsum += __shfl_xor(nsum, 16, 32);
    if (sel == 0) {
      denP[(size_t)tm * H_DIM + h] = dsum;
      numP[(size_t)tm * H_DIM + h] = nsum;
    }
  }
}

// --- final: 64-way partial reduction + LSTM candidate mix --------------------
__global__ void __launch_bounds__(256) k_final(
    const float* __restrict__ gates_pre, const float* __restrict__ denP,
    const float* __restrict__ numP, float* __restrict__ out) {
  int t = blockIdx.x * blockDim.x + threadIdx.x;
  if (t >= H_DIM) return;
  float ia = sigmoid_fast(gates_pre[t]);              // i
  float oa = sigmoid_fast(gates_pre[H_DIM + t]);      // o
  float ga = tanhf(gates_pre[2 * H_DIM + t]);         // g
  float wi = __expf(ia);
  float den = wi;
  float num = wi * ga;
#pragma unroll 4
  for (int m = 0; m < M_TILES; ++m) {
    den += denP[(size_t)m * H_DIM + t];
    num += numP[(size_t)m * H_DIM + t];
  }
  float c1 = num / den;
  out[t]         = oa * tanhf(c1);   // h1
  out[H_DIM + t] = c1;               // c1
}

extern "C" void kernel_launch(void* const* d_in, const int* in_sizes, int n_in,
                              void* d_out, int out_size, void* d_ws, size_t ws_size,
                              hipStream_t stream) {
  const float* x      = (const float*)d_in[0];
  const float* c_in   = (const float*)d_in[1];
  const float* h0     = (const float*)d_in[2];
  // d_in[3] = c0, unused by the reference math
  const float* W_ih   = (const float*)d_in[4];
  const float* b_ih   = (const float*)d_in[5];
  const float* W_hh   = (const float*)d_in[6];
  const float* b_hh   = (const float*)d_in[7];
  const float* aW_ih  = (const float*)d_in[8];
  const float* ab_ih  = (const float*)d_in[9];
  const float* aW_hh  = (const float*)d_in[10];
  const float* ab_hh  = (const float*)d_in[11];
  float* out = (float*)d_out;

  // workspace layout (bytes): bf16 [C*K + H*K] | gates[3H] | ax[H] |
  //                           denP[64*H] | numP[64*H]   (~26.3 MB total)
  char* ws = (char*)d_ws;
  unsigned short* bf16buf = (unsigned short*)ws;
  size_t off = (size_t)(C_DIM * (size_t)KDIM + H_DIM * (size_t)KDIM) * 2;
  float* gates_pre = (float*)(ws + off);  off += (size_t)3 * H_DIM * 4;
  float* ax        = (float*)(ws + off);  off += (size_t)H_DIM * 4;
  float* denP      = (float*)(ws + off);  off += (size_t)M_TILES * H_DIM * 4;
  float* numP      = (float*)(ws + off);

  const __bf16* Abf = (const __bf16*)bf16buf;                       // c_input bf16
  const __bf16* Bbf = Abf + (size_t)C_DIM * KDIM;                   // aW_hh  bf16

  // 1) fp32 -> bf16 pack of both GEMM operands
  {
    int n4 = (C_DIM * KDIM + H_DIM * KDIM) / 4;                     // 3,145,728
    k_convert_bf16<<<n4 / 256, 256, 0, stream>>>(c_in, aW_hh,
                                                 (unsigned int*)bf16buf);
  }
  // 2) LSTM gate GEMV (i, o, g): 6144 waves
  k_gates_gemv<<<(3 * H_DIM * 32) / 256, 256, 0, stream>>>(
      W_ih, b_ih, W_hh, b_hh, x, h0, gates_pre);
  // 3) ax GEMV: 2048 waves
  k_ax_gemv<<<(H_DIM * 32) / 256, 256, 0, stream>>>(aW_ih, ab_ih, ab_hh, x, ax);
  // 4) bf16 WMMA GEMM + fused softmax partials: 2048 waves (64x32 tiles)
  k_alpha_wmma<<<(M_TILES * (H_DIM / 64) * 32) / 256, 256, 0, stream>>>(
      Abf, Bbf, c_in, ax, denP, numP);
  // 5) final reduction + pointwise LSTM combine
  k_final<<<(H_DIM + 255) / 256, 256, 0, stream>>>(gates_pre, denP, numP, out);
}
